// InterventionalGraph_46196668236076
// MI455X (gfx1250) — matrix operand order repaired and checked by reading, not compiled, not run on hardware
//
#include <hip/hip_runtime.h>
#include <hip/hip_bf16.h>
#include <math.h>

typedef _Float16 h16;
typedef __attribute__((ext_vector_type(16))) _Float16 v16h;
typedef __attribute__((ext_vector_type(8)))  _Float16 v8h;
typedef __attribute__((ext_vector_type(8)))  float    v8f;

#define BS   128
#define T    200
#define LAG  30
#define TOUT (T - LAG)
#define NSK  1000
#define E    64
#define AH   64     // hidden width A
#define FE   128    // 2E edge-feature width
#define WPB  2      // waves per block

__device__ __forceinline__ float lrelu(float x) { return x >= 0.f ? x : 0.01f * x; }

// Wave-private LDS ordering: wait for outstanding DS ops (cross-lane store->load
// within one wave; no workgroup barrier needed for per-wave buffers).
__device__ __forceinline__ void wave_lds_fence() {
  asm volatile("s_wait_dscnt 0" ::: "memory");
}

#define WMMA(A, B, C) __builtin_amdgcn_wmma_f32_16x16x32_f16( \
    false, (A), false, (B), (short)0, (C), false, false)

// ---- Kernel 0: convert edge-MLP weights f32 -> f16 (row-major) into d_ws ----
// layout: [0 .. 8191] ie_w0 (64x128), [8192 .. 12287] ie_w1 (64x64),
//         [12288 .. 16383] ie_wo (64x64)
__global__ void wconvert_kernel(const float* __restrict__ w0,
                                const float* __restrict__ w1,
                                const float* __restrict__ wo,
                                h16* __restrict__ out) {
  int i = blockIdx.x * 256 + threadIdx.x;
  if (i < 8192)       out[i] = (h16)w0[i];
  else if (i < 12288) out[i] = (h16)w1[i - 8192];
  else if (i < 16384) out[i] = (h16)wo[i - 12288];
}

// ---- Main kernel: one wave32 per (b, t) target ----
__global__ __launch_bounds__(WPB * 32)
void ig_edge_kernel(const int* __restrict__ skills, const int* __restrict__ times,
                    const int* __restrict__ labels, const float* __restrict__ W,
                    const float* __restrict__ skill_base, const float* __restrict__ emb,
                    const float* __restrict__ ie_b0, const float* __restrict__ ie_b1,
                    const float* __restrict__ ie_bo, const h16* __restrict__ wpack,
                    const float* __restrict__ f_w0, const float* __restrict__ f_b0,
                    const float* __restrict__ f_w1, const float* __restrict__ f_b1,
                    const float* __restrict__ f_wo, const float* __restrict__ f_bo,
                    float* __restrict__ out) {
  __shared__ h16 w0h[AH * FE];           // 16 KB
  __shared__ h16 w1h[AH * AH];           //  8 KB
  __shared__ h16 woh[E * AH];            //  8 KB
  __shared__ float b0s[AH], b1s[AH], bos[E];
  __shared__ h16 scr[WPB][32 * FE];      // per-wave: feat (32x128) then H/H2 (each 32x64)
  __shared__ float coefs[WPB][32];
  __shared__ float cebuf[WPB][E];
  __shared__ float fvec[WPB][AH];

  const int tid = threadIdx.x;
  for (int i = tid; i < AH * FE; i += WPB * 32) w0h[i] = wpack[i];
  for (int i = tid; i < AH * AH; i += WPB * 32) w1h[i] = wpack[AH * FE + i];
  for (int i = tid; i < E * AH;  i += WPB * 32) woh[i] = wpack[AH * FE + AH * AH + i];
  for (int i = tid; i < AH; i += WPB * 32) { b0s[i] = ie_b0[i]; b1s[i] = ie_b1[i]; }
  for (int i = tid; i < E;  i += WPB * 32) bos[i] = ie_bo[i];
  __syncthreads();

  const int wv   = tid >> 5;
  const int lane = tid & 31;
  const int wid  = blockIdx.x * WPB + wv;
  const int b = wid / TOUT;
  const int t = wid % TOUT;

  const int   tgt = skills[b * T + t + LAG];
  const float tt  = (float)times[b * T + t + LAG];
  __builtin_prefetch(&emb[tgt * E], 0, 3);

  h16*   feat = &scr[wv][0];
  float* coef = &coefs[wv][0];

  // ---- Step 1: build 32x128 edge-feature tile (rows 30,31 zero) + coefficients ----
  {
    const int j = lane;
    if (j < LAG) {
      const int   sh   = skills[b * T + t + j];
      const int   lab  = labels[b * T + t + j];
      const float labf = (lab == 0) ? -1.f : ((lab == -1) ? 0.f : (float)lab);
      const float th   = (float)times[b * T + t + j];
      const float* et = &emb[tgt * E];
      const float* eh = &emb[sh * E];
      for (int k = 0; k < E; ++k) {
        feat[j * FE + k]     = (h16)et[k];
        feat[j * FE + E + k] = (h16)(eh[k] * labf);
      }
      const float wadj = (sh == tgt) ? 0.f : W[sh * NSK + tgt];
      const float inv_log5 = 0.6213349345596119f;  // 1/ln(5)
      const float delta = logf(fabsf(tt - th) + 1e-10f) * inv_log5;
      coef[j] = wadj * expf(-delta);
    } else {
      for (int k = 0; k < FE; ++k) feat[j * FE + k] = (h16)0.f;
      coef[j] = 0.f;
    }
  }
  wave_lds_fence();

  const int hiHalf = (lane >= 16) ? 8 : 0;   // A-layout K sub-chunk select
  const int arow   = (lane & 15);
  const int cn     = (lane & 15);            // C-layout column within tile
  const int cmBase = (lane >= 16) ? 8 : 0;   // C-layout row base

  // A fragment: 16x32 f16 tile, lanes 0-15 rows M, K = 0..15; lanes 16-31 K = 16..31
  auto load_a = [&](const h16* base, int stride, int row, int kt) -> v16h {
    v8h lo = *(const v8h*)&base[row * stride + kt * 32 + 2 * hiHalf];
    v8h hi = *(const v8h*)&base[row * stride + kt * 32 + 2 * hiHalf + 8];
    v16h a;
    for (int i = 0; i < 8; ++i) { a[i] = lo[i]; a[8 + i] = hi[i]; }
    return a;
  };
  // B fragment: 32x16 f16 tile from row-major weights w[n,k]; lanes 0-15 K=0..15, 16-31 K=16..31
  auto load_b = [&](const h16* wmat, int rowlen, int nt, int kt) -> v16h {
    const int n  = nt * 16 + (lane & 15);
    const int kb = kt * 32 + ((lane >= 16) ? 16 : 0);
    return *(const v16h*)&wmat[n * rowlen + kb];
  };

  h16* Hb  = &scr[wv][0];        // 32x64 hidden (overwrites feat after A-frags consumed)
  h16* H2b = &scr[wv][32 * AH];  // 32x64 hidden (post-residual)

  // ---- Step 2: layer0  H = lrelu(X @ W0^T + b0)   [32x128]x[128x64] ----
  v16h a0[2][4];
  for (int mt = 0; mt < 2; ++mt)
    for (int kt = 0; kt < 4; ++kt)
      a0[mt][kt] = load_a(feat, FE, arow + mt * 16, kt);

  for (int nt = 0; nt < 4; ++nt) {
    v8f c0 = {}, c1 = {};
    for (int kt = 0; kt < 4; ++kt) {
      v16h bf = load_b(w0h, FE, nt, kt);
      c0 = WMMA(a0[0][kt], bf, c0);
      c1 = WMMA(a0[1][kt], bf, c1);
    }
    const int n = nt * 16 + cn;
    const float bias = b0s[n];
    for (int r = 0; r < 8; ++r) {
      Hb[(cmBase + r) * AH + n]      = (h16)lrelu(c0[r] + bias);
      Hb[(16 + cmBase + r) * AH + n] = (h16)lrelu(c1[r] + bias);
    }
  }
  wave_lds_fence();

  // ---- Step 3: layer1  H2 = H + lrelu(H @ W1^T + b1) ----
  v16h a1[2][2];
  for (int mt = 0; mt < 2; ++mt)
    for (int kt = 0; kt < 2; ++kt)
      a1[mt][kt] = load_a(Hb, AH, arow + mt * 16, kt);

  for (int nt = 0; nt < 4; ++nt) {
    v8f c0 = {}, c1 = {};
    for (int kt = 0; kt < 2; ++kt) {
      v16h bf = load_b(w1h, AH, nt, kt);
      c0 = WMMA(a1[0][kt], bf, c0);
      c1 = WMMA(a1[1][kt], bf, c1);
    }
    const int n = nt * 16 + cn;
    const float bias = b1s[n];
    for (int r = 0; r < 8; ++r) {
      const int m0 = cmBase + r, m1 = 16 + cmBase + r;
      H2b[m0 * AH + n] = (h16)((float)Hb[m0 * AH + n] + lrelu(c0[r] + bias));
      H2b[m1 * AH + n] = (h16)((float)Hb[m1 * AH + n] + lrelu(c1[r] + bias));
    }
  }
  wave_lds_fence();

  // ---- Step 4: layerO msg = H2 @ Wo^T + bo, fused weighted reduction over edges ----
  v16h a2[2][2];
  for (int mt = 0; mt < 2; ++mt)
    for (int kt = 0; kt < 2; ++kt)
      a2[mt][kt] = load_a(H2b, AH, arow + mt * 16, kt);

  float cml[8], cmh[8];
  for (int r = 0; r < 8; ++r) {
    cml[r] = coef[cmBase + r];
    cmh[r] = coef[16 + cmBase + r];
  }
  const float sbase = skill_base[tgt];

  for (int nt = 0; nt < 4; ++nt) {
    v8f c0 = {}, c1 = {};
    for (int kt = 0; kt < 2; ++kt) {
      v16h bf = load_b(woh, AH, nt, kt);
      c0 = WMMA(a2[0][kt], bf, c0);
      c1 = WMMA(a2[1][kt], bf, c1);
    }
    const int n = nt * 16 + cn;
    const float bias = bos[n];
    float part = 0.f;
    for (int r = 0; r < 8; ++r)
      part += cml[r] * (c0[r] + bias) + cmh[r] * (c1[r] + bias);
    part += __shfl_xor(part, 16, 32);
    if (lane < 16) cebuf[wv][n] = part + sbase;
  }
  wave_lds_fence();

  // ---- Step 5: final MLP (64->64->64->1) + sigmoid, VALU (cheap) ----
  const float* x = &cebuf[wv][0];
  const int n1 = lane, n2 = lane + 32;
  float acc1 = f_b0[n1], acc2 = f_b0[n2];
  for (int k = 0; k < E; ++k) {
    const float xv = x[k];
    acc1 += f_w0[n1 * E + k] * xv;
    acc2 += f_w0[n2 * E + k] * xv;
  }
  const float h1  = x[n1] + lrelu(acc1);
  const float h2v = x[n2] + lrelu(acc2);
  fvec[wv][n1] = h1;
  fvec[wv][n2] = h2v;
  wave_lds_fence();

  float g1 = f_b1[n1], g2 = f_b1[n2];
  for (int k = 0; k < AH; ++k) {
    const float hv = fvec[wv][k];
    g1 += f_w1[n1 * AH + k] * hv;
    g2 += f_w1[n2 * AH + k] * hv;
  }
  const float hh1 = h1 + lrelu(g1);
  const float hh2 = h2v + lrelu(g2);
  float part = f_wo[n1] * hh1 + f_wo[n2] * hh2;
  for (int s = 16; s > 0; s >>= 1) part += __shfl_xor(part, s, 32);
  if (lane == 0) {
    const float o = part + f_bo[0];
    out[b * TOUT + t] = 1.f / (1.f + expf(-o));
  }
}

extern "C" void kernel_launch(void* const* d_in, const int* in_sizes, int n_in,
                              void* d_out, int out_size, void* d_ws, size_t ws_size,
                              hipStream_t stream) {
  const int*   skills = (const int*)d_in[0];
  const int*   times  = (const int*)d_in[1];
  const int*   labels = (const int*)d_in[2];
  const float* W      = (const float*)d_in[3];
  const float* sbase  = (const float*)d_in[4];
  const float* emb    = (const float*)d_in[5];
  const float* ie_w0  = (const float*)d_in[6];
  const float* ie_b0  = (const float*)d_in[7];
  const float* ie_w1  = (const float*)d_in[8];
  const float* ie_b1  = (const float*)d_in[9];
  const float* ie_wo  = (const float*)d_in[10];
  const float* ie_bo  = (const float*)d_in[11];
  const float* f_w0   = (const float*)d_in[12];
  const float* f_b0   = (const float*)d_in[13];
  const float* f_w1   = (const float*)d_in[14];
  const float* f_b1   = (const float*)d_in[15];
  const float* f_wo   = (const float*)d_in[16];
  const float* f_bo   = (const float*)d_in[17];
  float* out = (float*)d_out;
  h16* wpack = (h16*)d_ws;

  hipLaunchKernelGGL(wconvert_kernel, dim3(64), dim3(256), 0, stream,
                     ie_w0, ie_w1, ie_wo, wpack);

  const int nblocks = (BS * TOUT) / WPB;  // one wave32 per (b, t)
  hipLaunchKernelGGL(ig_edge_kernel, dim3(nblocks), dim3(WPB * 32), 0, stream,
                     skills, times, labels, W, sbase, emb, ie_b0, ie_b1, ie_bo, wpack,
                     f_w0, f_b0, f_w1, f_b1, f_wo, f_bo, out);
}
